// AdditiveAttn_79319456023175
// MI455X (gfx1250) — compile-verified, matrix operand order checked by reading, not run on hardware
//
#include <hip/hip_runtime.h>
#include <hip/hip_bf16.h>
#include <math.h>

#define HID    1024
#define TSTEP  2048
#define BATCH  32
#define K2H    2048
#define KTILES 32          // 1024 / 32
#define NTILES 64          // 1024 / 16
#define FRAGE  16          // bf16 elements per lane per fragment
#define W1P_ELEMS ((size_t)NTILES * KTILES * 32 * FRAGE)   // 1,048,576 (2 MB)

typedef __attribute__((ext_vector_type(16))) __bf16 v16bf;
typedef __attribute__((ext_vector_type(4)))  __bf16 v4bf;
typedef __attribute__((ext_vector_type(8)))  float  v8f;

static __device__ inline void cvt4(v16bf& dst, int base, float4 v) {
    dst[base + 0] = (__bf16)v.x;
    dst[base + 1] = (__bf16)v.y;
    dst[base + 2] = (__bf16)v.z;
    dst[base + 3] = (__bf16)v.w;
}

// Branch-free tanh: 1 - 2/(exp(2x)+1).  v_exp_f32 + v_rcp_f32, saturates via inf.
static __device__ inline float fast_tanh(float x) {
    float e = __builtin_amdgcn_exp2f(x * 2.8853900817779268f);  // exp(2x)
    float r = __builtin_amdgcn_rcpf(e + 1.0f);
    return fmaf(-2.0f, r, 1.0f);
}

// ---------------------------------------------------------------------------
// Kernel 0: pack W1[:, 0:1024] (enc half) fp32 -> bf16 in B-fragment order.
// Fragment (nt, kt): lane holds n = nt*16+(lane&15), khalf = (lane>>4)*8,
// elements e0..7 = K k0+khalf.., e8..15 = K k0+16+khalf..  (ISA 16-bit B layout)
// Storage: w1p[((nt*KTILES+kt)*32 + lane)*16 + e]  -> 32B/lane, coalesced.
// ---------------------------------------------------------------------------
__global__ __launch_bounds__(256)
void w1pack_kernel(const float* __restrict__ W1, __bf16* __restrict__ w1p)
{
    const int gid  = blockIdx.x * 256 + threadIdx.x;   // 0 .. 65535
    const int lane = gid & 31;
    const int kt   = (gid >> 5) & 31;
    const int nt   = gid >> 10;
    const int n    = nt * 16 + (lane & 15);
    const int k0   = kt * 32 + ((lane >> 4) * 8);
    const float* __restrict__ src = W1 + (size_t)n * K2H + k0;
    v16bf b;
    cvt4(b, 0,  *(const float4*)(src));
    cvt4(b, 4,  *(const float4*)(src + 4));
    cvt4(b, 8,  *(const float4*)(src + 16));
    cvt4(b, 12, *(const float4*)(src + 20));
    *(v16bf*)(w1p + (size_t)gid * FRAGE) = b;
}

// ---------------------------------------------------------------------------
// Kernel 1: ybias[b,h] = cur[b,:] . W1[h, 1024:2048] + b1[h]
// ---------------------------------------------------------------------------
__global__ __launch_bounds__(256)
void ybias_kernel(const float* __restrict__ cur,
                  const float* __restrict__ W1,
                  const float* __restrict__ b1,
                  float* __restrict__ ybias)
{
    const int tid = threadIdx.x;
    const int b   = tid & 31;
    const int hl  = tid >> 5;
    const int h   = blockIdx.x * 8 + hl;      // grid = 128 blocks
    const float* __restrict__ wrow = W1 + (size_t)h * K2H + HID;
    const float* __restrict__ crow = cur + (size_t)b * HID;
    float acc = 0.f;
    #pragma unroll 4
    for (int k = 0; k < HID; ++k)
        acc = fmaf(crow[k], wrow[k], acc);
    ybias[b * HID + h] = acc + b1[h];
}

// ---------------------------------------------------------------------------
// Kernel 2: scores[r] = W2 . tanh(enc_row . W1a^T + ybias[b]) + b2
// Block = 4 waves = 16 rows of flattened M = t*B + b.
// A tile staged via GLOBAL_LOAD_ASYNC_TO_LDS_B128 (ASYNCcnt), converted to
// fragment-ordered bf16 LDS; B fragments streamed pre-packed from L2.
// ---------------------------------------------------------------------------
__global__ __launch_bounds__(128)
void attn_scores_kernel(const float* __restrict__ enc,
                        const __bf16* __restrict__ w1p,
                        const float* __restrict__ W2,
                        const float* __restrict__ ybias,
                        const float* __restrict__ b2,
                        float* __restrict__ scores)
{
    __shared__ float  Af[4 * HID];                    // 16 KB async staging chunk
    __shared__ __bf16 Asf[KTILES * 32 * FRAGE];       // 32 KB, fragment order
    __shared__ float  sred[16];

    const int tid   = threadIdx.x;        // 0..127
    const int wave  = tid >> 5;           // 0..3
    const int lane  = tid & 31;
    const int mbase = blockIdx.x * 16;    // row base in flattened (t*B + b)

    if (tid < 16) sred[tid] = 0.0f;

    // ---- stage A tile: 16 x 1024 fp32, 4 chunks of 4 rows via async-to-LDS ----
    const float* encBase = enc + (size_t)mbase * HID;
    for (int c = 0; c < 4; ++c) {
        #pragma unroll
        for (int j = 0; j < 8; ++j) {
            const int idx = j * 128 + tid;                       // vec4 idx in chunk
            unsigned ldsoff = (unsigned)(uintptr_t)(&Af[idx * 4]);
            unsigned long long ga =
                (unsigned long long)(uintptr_t)(encBase + c * (4 * HID) + idx * 4);
            asm volatile("global_load_async_to_lds_b128 %0, %1, off"
                         :: "v"(ldsoff), "v"(ga) : "memory");
        }
        asm volatile("s_wait_asynccnt 0" ::: "memory");
        __syncthreads();
        // convert chunk fp32 -> bf16 into fragment-ordered Asf (ds_store_b64)
        for (int idx = tid; idx < (4 * HID) / 4; idx += 128) {
            float4 v = *(const float4*)(&Af[idx * 4]);
            const int m  = c * 4 + (idx >> 8);       // row 0..15
            const int k  = (idx & 255) * 4;          // col, multiple of 4
            const int kt = k >> 5;
            const int kk = k & 31;
            const int ld = m + (((kk >> 3) & 1) << 4);       // dest lane
            const int e0 = (kk & 7) + ((kk >> 4) << 3);      // dest element (mult of 4)
            v4bf p;
            p[0] = (__bf16)v.x; p[1] = (__bf16)v.y;
            p[2] = (__bf16)v.z; p[3] = (__bf16)v.w;
            *(v4bf*)(&Asf[((kt * 32 + ld) << 4) + e0]) = p;
        }
        __syncthreads();
    }

    const int bofm = mbase & (BATCH - 1);   // batch index of m=0 (MTILE=16 | B=32)
    const int moff = (lane >> 4) * 8;       // C-matrix: lanes>=16 hold M=j+8

    float partial[8];
    #pragma unroll
    for (int j = 0; j < 8; ++j) partial[j] = 0.f;

    for (int nt = wave; nt < NTILES; nt += 4) {
        const int n = nt * 16 + (lane & 15);          // output column h for this lane
        const __bf16* __restrict__ bfrag =
            w1p + ((size_t)(nt * KTILES) * 32 + lane) * FRAGE;

        v8f acc = {};
        #pragma unroll 4
        for (int kt = 0; kt < KTILES; ++kt) {
            v16bf a = *(const v16bf*)(&Asf[(kt * 32 + lane) << 4]);     // 2x ds_load_b128
            v16bf b = *(const v16bf*)(bfrag + (size_t)kt * 32 * FRAGE); // 2x global_load_b128
            if ((kt & 3) == 0)
                __builtin_prefetch(bfrag + (size_t)(kt + 4) * 32 * FRAGE, 0, 1);
            acc = __builtin_amdgcn_wmma_f32_16x16x32_bf16(
                      /*neg_a=*/false, a, /*neg_b=*/false, b,
                      /*c_mod=*/(short)0, acc,
                      /*reuse_a=*/false, /*reuse_b=*/false);
        }

        // ---- epilogue: branch-free tanh + W2-weighted reduction over n ----
        const float w2n = W2[n];
        const float* __restrict__ yb = ybias + (size_t)(bofm + moff) * HID + n;
        #pragma unroll
        for (int j = 0; j < 8; ++j) {
            const float y = acc[j] + yb[(size_t)j * HID];
            partial[j] = fmaf(fast_tanh(y), w2n, partial[j]);
        }
    }

    // reduce partial scores across lanes/waves via LDS atomics (DS_ADD_F32)
    #pragma unroll
    for (int j = 0; j < 8; ++j)
        atomicAdd(&sred[j + moff], partial[j]);
    __syncthreads();

    if (tid < 16)
        scores[mbase + tid] = sred[tid] + b2[0];
}

// ---------------------------------------------------------------------------
// Kernel 3: softmax over T for each b; writes attn (B,T) into d_out.
// ---------------------------------------------------------------------------
__global__ __launch_bounds__(256)
void softmax_kernel(const float* __restrict__ scores,
                    float* __restrict__ attn)
{
    const int b   = blockIdx.x;     // 32 blocks
    const int tid = threadIdx.x;
    __shared__ float red[256];

    float mx = -INFINITY;
    for (int t = tid; t < TSTEP; t += 256)
        mx = fmaxf(mx, scores[t * BATCH + b]);
    red[tid] = mx; __syncthreads();
    for (int s = 128; s > 0; s >>= 1) {
        if (tid < s) red[tid] = fmaxf(red[tid], red[tid + s]);
        __syncthreads();
    }
    mx = red[0];
    __syncthreads();

    float sum = 0.f;
    for (int t = tid; t < TSTEP; t += 256)
        sum += __expf(scores[t * BATCH + b] - mx);
    red[tid] = sum; __syncthreads();
    for (int s = 128; s > 0; s >>= 1) {
        if (tid < s) red[tid] += red[tid + s];
        __syncthreads();
    }
    const float inv = 1.0f / red[0];

    for (int t = tid; t < TSTEP; t += 256)
        attn[b * TSTEP + t] = __expf(scores[t * BATCH + b] - mx) * inv;
}

// ---------------------------------------------------------------------------
// Kernel 4: context partials over T splits (deterministic, no float atomics)
// ---------------------------------------------------------------------------
__global__ __launch_bounds__(256)
void ctx_partial_kernel(const float* __restrict__ enc,
                        const float* __restrict__ attn,
                        float* __restrict__ part)
{
    const int h  = blockIdx.x * 256 + threadIdx.x;   // gridDim.x = 4
    const int ts = blockIdx.y;                       // gridDim.y = 8
    const int b  = blockIdx.z;                       // gridDim.z = 32
    const int t0 = ts * (TSTEP / 8);

    float acc = 0.f;
    for (int t = t0; t < t0 + TSTEP / 8; ++t)
        acc = fmaf(attn[b * TSTEP + t],
                   enc[((size_t)t * BATCH + b) * HID + h], acc);
    part[(size_t)ts * (BATCH * HID) + b * HID + h] = acc;
}

// ---------------------------------------------------------------------------
// Kernel 5: reduce the 8 T-splits -> cntxt (B,H) at d_out[0]
// ---------------------------------------------------------------------------
__global__ __launch_bounds__(256)
void ctx_reduce_kernel(const float* __restrict__ part,
                       float* __restrict__ cntxt)
{
    const int i = blockIdx.x * 256 + threadIdx.x;    // 0 .. 32767
    float acc = 0.f;
    #pragma unroll
    for (int ts = 0; ts < 8; ++ts)
        acc += part[(size_t)ts * (BATCH * HID) + i];
    cntxt[i] = acc;
}

// ---------------------------------------------------------------------------
extern "C" void kernel_launch(void* const* d_in, const int* in_sizes, int n_in,
                              void* d_out, int out_size, void* d_ws, size_t ws_size,
                              hipStream_t stream) {
    (void)in_sizes; (void)n_in; (void)out_size; (void)ws_size;

    const float* enc = (const float*)d_in[0];   // (T, B, H)
    const float* cur = (const float*)d_in[1];   // (B, H)
    const float* W1  = (const float*)d_in[2];   // (H, 2H)
    const float* b1  = (const float*)d_in[3];   // (H,)
    const float* W2  = (const float*)d_in[4];   // (1, H)
    const float* b2  = (const float*)d_in[5];   // (1,)

    float* out   = (float*)d_out;
    float* cntxt = out;                          // (B, H)
    float* attn  = out + BATCH * HID;            // (B, T)

    __bf16* w1p  = (__bf16*)d_ws;                // 2 MB packed bf16 W1 (enc half)
    float* fws    = (float*)((char*)d_ws + W1P_ELEMS * sizeof(__bf16));
    float* scores = fws;                         // T*B   = 65536 floats
    float* ybias  = fws + TSTEP * BATCH;         // B*H   = 32768 floats
    float* part   = ybias + BATCH * HID;         // 8*B*H = 262144 floats

    w1pack_kernel<<<(NTILES * KTILES * 32) / 256, 256, 0, stream>>>(W1, w1p);
    ybias_kernel<<<HID / 8, 256, 0, stream>>>(cur, W1, b1, ybias);
    attn_scores_kernel<<<(TSTEP * BATCH) / 16, 128, 0, stream>>>(
        enc, w1p, W2, ybias, b2, scores);
    softmax_kernel<<<BATCH, 256, 0, stream>>>(scores, attn);
    ctx_partial_kernel<<<dim3(HID / 256, 8, BATCH), 256, 0, stream>>>(
        enc, attn, part);
    ctx_reduce_kernel<<<(BATCH * HID) / 256, 256, 0, stream>>>(part, cntxt);
}
